// MambaBlock_30863634989742
// MI455X (gfx1250) — matrix operand order repaired and checked by reading, not compiled, not run on hardware
//
#include <hip/hip_runtime.h>
#include <hip/hip_bf16.h>
#include <cstdint>

#define B_  8
#define L_  1024
#define D_  512
#define R_  32
#define N_  16
#define M_TOT (B_ * L_)   // 8192

typedef __attribute__((ext_vector_type(16))) __bf16 v16bf;
typedef __attribute__((ext_vector_type(8)))  float  v8f;
typedef __attribute__((ext_vector_type(4)))  int    v4i;

union FragU { v16bf v; uint4 q[2]; };

// Prefer the gfx1250 async global->LDS path when the toolchain exposes it.
#if defined(__has_builtin)
#if __has_builtin(__builtin_amdgcn_global_load_async_to_lds_b128) && \
    __has_builtin(__builtin_amdgcn_s_wait_asynccnt)
#define USE_ASYNC_LDS 1
#endif
#endif
#ifndef USE_ASYNC_LDS
#define USE_ASYNC_LDS 0
#endif

#if USE_ASYNC_LDS
typedef __attribute__((address_space(1))) v4i* gv4i_p;   // global
typedef __attribute__((address_space(3))) v4i* lv4i_p;   // LDS
#endif

__device__ __forceinline__ void cp16_g2l(void* lds, const void* gsrc) {
#if USE_ASYNC_LDS
  __builtin_amdgcn_global_load_async_to_lds_b128((gv4i_p)gsrc, (lv4i_p)lds, 0, 0);
#else
  *(uint4*)lds = *(const uint4*)gsrc;
#endif
}
__device__ __forceinline__ void cp_wait() {
#if USE_ASYNC_LDS
  __builtin_amdgcn_s_wait_asynccnt(0);
#endif
}

// D = A(16x32) * B(32x16) + C, fp32 accumulate  -> v_wmma_f32_16x16x32_bf16
__device__ __forceinline__ v8f wmma_bf16(v16bf a, v16bf b, v8f c) {
  return __builtin_amdgcn_wmma_f32_16x16x32_bf16(false, a, false, b, (short)0, c,
                                                 false, false);
}

// A fragment (16x32, row-major in LDS, row stride ld elements).
// ISA layout: lane m(0..15) holds K=0..7 & 16..23; lane m+16 holds K=8..15 & 24..31.
__device__ __forceinline__ v16bf load_frag_a(const __bf16* base, int ld) {
  const int lane = threadIdx.x & 31;
  const int m = lane & 15, hi = lane >> 4;
  const __bf16* p = base + m * ld + hi * 8;
  FragU f;
  f.q[0] = *(const uint4*)(p);
  f.q[1] = *(const uint4*)(p + 16);
  return f.v;
}

// B fragment from Bt (N x K row-major, stride ld): lane n(0..15) holds K=0..7 & 16..23
// of column n, lane n+16 holds K=8..15 & 24..31 (matches A layout).
__device__ __forceinline__ v16bf load_frag_b(const __bf16* base, int ld) {
  const int lane = threadIdx.x & 31;
  const int n = lane & 15, hi = lane >> 4;
  const __bf16* p = base + n * ld + hi * 8;
  FragU f;
  f.q[0] = *(const uint4*)(p);
  f.q[1] = *(const uint4*)(p + 16);
  return f.v;
}

__device__ __forceinline__ float silu_f(float v)     { return v / (1.f + __expf(-v)); }
__device__ __forceinline__ float softplus_f(float v) { return v > 20.f ? v : log1pf(__expf(v)); }

// ---------------------------------------------------------------- converts ---
__global__ void cvt_bf16_kernel(const float* __restrict__ in,
                                __bf16* __restrict__ out, int n) {
  int i = blockIdx.x * blockDim.x + threadIdx.x;
  int stride = gridDim.x * blockDim.x;
  for (; i < n; i += stride) out[i] = (__bf16)in[i];
}

// conv_w (L,L,3) fp32 -> 3 contiguous (L,L) bf16 tap matrices
__global__ void pack_convw_kernel(const float* __restrict__ cw,
                                  __bf16* __restrict__ out) {
  size_t i = (size_t)blockIdx.x * 256 + threadIdx.x;   // over L_*L_
  const float* s = cw + i * 3;
  out[i]                       = (__bf16)s[0];
  out[(size_t)L_ * L_ + i]     = (__bf16)s[1];
  out[(size_t)2 * L_ * L_ + i] = (__bf16)s[2];
}

// ------------------------------------------------------------ generic GEMM ---
// C = A * Bt^T ; A is M x K (bf16, or fp32 converted on the fly when ACVT),
// Bt is N x K row-major bf16.  Grid: (N/BN, M/BM), 256 threads = 8 waves.
// Double-buffered LDS; next tile staged (async where available) before compute.
// MODE 0: +bias[n], store bf16 transposed-per-batch  (x1t[b][n][l])
// MODE 1: +bias[n], silu, store fp32
// MODE 2: store fp32
// MODE 3: +bias[n], store fp32
// MODE 4: +bias[n], softplus, store fp32
template<int BM, int BN, int WGM, int WGN, int MODE, bool ACVT>
__global__ __launch_bounds__(256) void gemm_kernel(
    const void* __restrict__ Aptr, int lda,
    const __bf16* __restrict__ Bt, int ldb,
    const float* __restrict__ bias,
    float* __restrict__ outf, __bf16* __restrict__ outb,
    int K, int ldc) {
  static_assert(WGM * WGN == 8, "8 waves");
  constexpr int MT = BM / 16 / WGM, NT = BN / 16 / WGN;
  constexpr int LDS_K = 40;  // 32 + pad
  __shared__ __align__(16) __bf16 As[2][BM][LDS_K];
  __shared__ __align__(16) __bf16 Bs[2][BN][LDS_K];

  const int m0 = blockIdx.y * BM, n0 = blockIdx.x * BN;
  const int w = threadIdx.x >> 5;
  const int wm = w / WGN, wn = w % WGN;

  v8f zero;
#pragma unroll
  for (int i = 0; i < 8; i++) zero[i] = 0.f;
  v8f acc[MT][NT];
#pragma unroll
  for (int mi = 0; mi < MT; mi++)
#pragma unroll
    for (int ni = 0; ni < NT; ni++) acc[mi][ni] = zero;

  auto stage = [&](int kt, int buf) {
    for (int c = threadIdx.x; c < BM * 4; c += 256) {
      int row = c >> 2, off = (c & 3) * 8;
      if constexpr (ACVT) {
        const float* s = (const float*)Aptr + (size_t)(m0 + row) * lda + kt + off;
        __bf16 tmp[8];
#pragma unroll
        for (int i = 0; i < 8; i++) tmp[i] = (__bf16)s[i];
        *(uint4*)&As[buf][row][off] = *(const uint4*)tmp;
      } else {
        cp16_g2l(&As[buf][row][off],
                 (const __bf16*)Aptr + (size_t)(m0 + row) * lda + kt + off);
      }
    }
    for (int c = threadIdx.x; c < BN * 4; c += 256) {
      int row = c >> 2, off = (c & 3) * 8;
      cp16_g2l(&Bs[buf][row][off], Bt + (size_t)(n0 + row) * ldb + kt + off);
    }
  };

  stage(0, 0);
  cp_wait();
  __syncthreads();
  int cur = 0;
  for (int kt = 0; kt < K; kt += 32) {
    if (kt + 32 < K) stage(kt + 32, cur ^ 1);  // overlaps with WMMAs below

    v16bf af[MT], bfr[NT];
#pragma unroll
    for (int mi = 0; mi < MT; mi++)
      af[mi] = load_frag_a(&As[cur][(wm * MT + mi) * 16][0], LDS_K);
#pragma unroll
    for (int ni = 0; ni < NT; ni++)
      bfr[ni] = load_frag_b(&Bs[cur][(wn * NT + ni) * 16][0], LDS_K);
#pragma unroll
    for (int mi = 0; mi < MT; mi++)
#pragma unroll
      for (int ni = 0; ni < NT; ni++)
        acc[mi][ni] = wmma_bf16(af[mi], bfr[ni], acc[mi][ni]);

    cp_wait();
    __syncthreads();
    cur ^= 1;
  }

  // epilogue (C/D layout: VGPR r -> row r + 8*hi, lane&15 -> col)
  const int lane = threadIdx.x & 31, col = lane & 15, hi = lane >> 4;
#pragma unroll
  for (int mi = 0; mi < MT; mi++) {
#pragma unroll
    for (int ni = 0; ni < NT; ni++) {
      const int gm = m0 + (wm * MT + mi) * 16 + hi * 8;
      const int gn = n0 + (wn * NT + ni) * 16 + col;
      if constexpr (MODE == 0) {
        const int bb = gm >> 10, l = gm & 1023;   // batch / position
        const float bv = bias[gn];
        __bf16 tmp[8];
#pragma unroll
        for (int r = 0; r < 8; r++) tmp[r] = (__bf16)(acc[mi][ni][r] + bv);
        *(uint4*)(outb + ((size_t)bb << 19) + (size_t)gn * L_ + l) = *(const uint4*)tmp;
      } else {
        const float bv = (MODE == 2) ? 0.f : bias[gn];
#pragma unroll
        for (int r = 0; r < 8; r++) {
          float v = acc[mi][ni][r] + bv;
          if constexpr (MODE == 1) v = silu_f(v);
          if constexpr (MODE == 4) v = softplus_f(v);
          outf[(size_t)(gm + r) * ldc + gn] = v;
        }
      }
    }
  }
}

// ----------------------------------------------------------- conv as GEMM ---
// out[b,lo,d] = silu( sum_li sum_k cw_k[lo,li] * x1t[b][d+k-1][li] + conv_b[lo] )
// Grid: (512/BN, 1024/BM, B_).  Bt tile carries a +/-1 row halo for the taps.
// Register-staged pipeline: next K-tile is loaded to VGPRs while the 24 WMMAs
// of the current tile run, then spilled to LDS (LDS too big to double-buffer).
__global__ __launch_bounds__(256) void conv_gemm_kernel(
    const __bf16* __restrict__ cw,    // 3 x (1024x1024)
    const __bf16* __restrict__ x1t,   // per batch 512x1024 (d-major)
    const float* __restrict__ conv_b,
    float* __restrict__ u, __bf16* __restrict__ ub) {
  constexpr int BM = 128, BN = 128, MT = 2, NT = 4, LDS_K = 40;
  __shared__ __align__(16) __bf16 As[3][BM][LDS_K];
  __shared__ __align__(16) __bf16 Bs[BN + 2][LDS_K];

  const int b = blockIdx.z;
  const int m0 = blockIdx.y * BM;   // l_o
  const int n0 = blockIdx.x * BN;   // d
  const __bf16* xb = x1t + (size_t)b * D_ * L_;
  const int w = threadIdx.x >> 5;
  const int wm = w >> 1, wn = w & 1;

  v8f zero;
#pragma unroll
  for (int i = 0; i < 8; i++) zero[i] = 0.f;
  v8f acc[MT][NT];
#pragma unroll
  for (int mi = 0; mi < MT; mi++)
#pragma unroll
    for (int ni = 0; ni < NT; ni++) acc[mi][ni] = zero;

  uint4 aR[6], bR[3];
  auto gload = [&](int kt) {
    int i = 0;
    for (int c = threadIdx.x; c < 3 * BM * 4; c += 256) {   // 6 chunks/thread
      int k = c / (BM * 4), rr = c - k * BM * 4;
      int row = rr >> 2, off = (rr & 3) * 8;
      const __bf16* s = cw + (size_t)k * L_ * L_ + (size_t)(m0 + row) * L_ + kt + off;
      __builtin_prefetch(s + 32, 0, 0);   // gfx1250 global_prefetch_b8 (next tile)
      aR[i++] = *(const uint4*)s;
    }
    i = 0;
    for (int c = threadIdx.x; c < (BN + 2) * 4; c += 256) {
      int row = c >> 2, off = (c & 3) * 8;
      int d = n0 + row - 1;                       // halo row 0 == d-1
      uint4 val = make_uint4(0u, 0u, 0u, 0u);     // zero-pad boundary
      if (d >= 0 && d < D_) val = *(const uint4*)(xb + (size_t)d * L_ + kt + off);
      bR[i++] = val;
    }
  };
  auto lstore = [&]() {
    int i = 0;
    for (int c = threadIdx.x; c < 3 * BM * 4; c += 256) {
      int k = c / (BM * 4), rr = c - k * BM * 4;
      int row = rr >> 2, off = (rr & 3) * 8;
      *(uint4*)&As[k][row][off] = aR[i++];
    }
    i = 0;
    for (int c = threadIdx.x; c < (BN + 2) * 4; c += 256) {
      int row = c >> 2, off = (c & 3) * 8;
      *(uint4*)&Bs[row][off] = bR[i++];
    }
  };

  gload(0);
  lstore();
  __syncthreads();
  for (int kt = 0; kt < L_; kt += 32) {   // K = l_i = 1024
    const bool more = (kt + 32) < L_;
    if (more) gload(kt + 32);             // overlaps with the tap WMMAs

#pragma unroll
    for (int k = 0; k < 3; k++) {   // taps
      v16bf af[MT], bfr[NT];
#pragma unroll
      for (int mi = 0; mi < MT; mi++)
        af[mi] = load_frag_a(&As[k][(wm * MT + mi) * 16][0], LDS_K);
#pragma unroll
      for (int ni = 0; ni < NT; ni++)
        bfr[ni] = load_frag_b(&Bs[(wn * NT + ni) * 16 + k][0], LDS_K);
#pragma unroll
      for (int mi = 0; mi < MT; mi++)
#pragma unroll
        for (int ni = 0; ni < NT; ni++)
          acc[mi][ni] = wmma_bf16(af[mi], bfr[ni], acc[mi][ni]);
    }

    if (more) {
      __syncthreads();   // all reads of current tile done
      lstore();
      __syncthreads();   // new tile visible to all waves
    }
  }

  const int lane = threadIdx.x & 31, col = lane & 15, hi = lane >> 4;
#pragma unroll
  for (int mi = 0; mi < MT; mi++) {
#pragma unroll
    for (int ni = 0; ni < NT; ni++) {
      const int gm = m0 + (wm * MT + mi) * 16 + hi * 8;
      const int gn = n0 + (wn * NT + ni) * 16 + col;
#pragma unroll
      for (int r = 0; r < 8; r++) {
        float v = acc[mi][ni][r] + conv_b[gm + r];  // bias over channel = l_o
        v = silu_f(v);
        size_t idx = ((size_t)b * L_ + gm + r) * D_ + gn;
        u[idx] = v;
        ub[idx] = (__bf16)v;
      }
    }
  }
}

// ---------------------------------------------------------- selective scan ---
// one thread per (b,d); 16 SSM states in registers; Bm/Cm staged in LDS in
// blocks of 8 timesteps (2 barriers per 8 steps instead of per step).
__global__ __launch_bounds__(128) void scan_kernel(
    const float* __restrict__ delta, const float* __restrict__ u,
    const float* __restrict__ dbc,   const float* __restrict__ x,
    const float* __restrict__ g2,    const float* __restrict__ A_log,
    const float* __restrict__ Dp,    __bf16* __restrict__ zb) {
  const int b = blockIdx.y;
  const int d = blockIdx.x * 128 + threadIdx.x;
  float a[N_], h[N_];
#pragma unroll
  for (int n = 0; n < N_; n++) {
    a[n] = -__expf(A_log[d * N_ + n]);
    h[n] = 0.f;
  }
  const float dpv = Dp[d];
  __shared__ float sBC[8][2 * N_];
  const size_t base = (size_t)b * L_ * D_ + d;
  const size_t dbcb = (size_t)b * L_ * (R_ + 2 * N_);

  for (int t0 = 0; t0 < L_; t0 += 8) {
    for (int i = threadIdx.x; i < 8 * 2 * N_; i += 128) {
      int tt = i >> 5, j = i & 31;
      sBC[tt][j] = dbc[dbcb + (size_t)(t0 + tt) * (R_ + 2 * N_) + R_ + j];
    }
    __syncthreads();
#pragma unroll
    for (int tt = 0; tt < 8; tt++) {
      const size_t idx = base + (size_t)(t0 + tt) * D_;
      const float dl = delta[idx], uv = u[idx];
      const float dlu = dl * uv;
      float y = 0.f;
#pragma unroll
      for (int n = 0; n < N_; n++) {
        float dA = __expf(dl * a[n]);
        h[n] = fmaf(dA, h[n], dlu * sBC[tt][n]);
        y = fmaf(h[n], sBC[tt][N_ + n], y);
      }
      float z = fmaf(uv, dpv, y) * g2[idx] + x[idx];
      zb[idx] = (__bf16)z;
    }
    __syncthreads();
  }
}

// ------------------------------------------------------------------ launch ---
extern "C" void kernel_launch(void* const* d_in, const int* in_sizes, int n_in,
                              void* d_out, int out_size, void* d_ws, size_t ws_size,
                              hipStream_t stream) {
  (void)in_sizes; (void)n_in; (void)out_size; (void)ws_size;
  const float* x      = (const float*)d_in[0];
  const float* W1     = (const float*)d_in[1];
  const float* b1     = (const float*)d_in[2];
  const float* W2     = (const float*)d_in[3];
  const float* b2     = (const float*)d_in[4];
  const float* Wout   = (const float*)d_in[5];
  const float* bout   = (const float*)d_in[6];
  const float* conv_w = (const float*)d_in[7];
  const float* conv_b = (const float*)d_in[8];
  const float* W_dbc  = (const float*)d_in[9];
  const float* W_dt   = (const float*)d_in[10];
  const float* b_dt   = (const float*)d_in[11];
  const float* A_log  = (const float*)d_in[12];
  const float* Dp     = (const float*)d_in[13];

  char* wp = (char*)d_ws;
  auto alloc = [&](size_t bytes) -> char* {
    char* p = wp;
    wp += (bytes + 255) & ~(size_t)255;
    return p;
  };
  __bf16* x_bf  = (__bf16*)alloc((size_t)M_TOT * D_ * 2);
  __bf16* w1b   = (__bf16*)alloc((size_t)D_ * D_ * 2);
  __bf16* w2b   = (__bf16*)alloc((size_t)D_ * D_ * 2);
  __bf16* wdbcb = (__bf16*)alloc((size_t)(R_ + 2 * N_) * D_ * 2);
  __bf16* wdtb  = (__bf16*)alloc((size_t)D_ * R_ * 2);
  __bf16* woutb = (__bf16*)alloc((size_t)D_ * D_ * 2);
  __bf16* cwb   = (__bf16*)alloc((size_t)3 * L_ * L_ * 2);
  __bf16* x1t   = (__bf16*)alloc((size_t)B_ * D_ * L_ * 2);
  float*  g2    = (float*) alloc((size_t)M_TOT * D_ * 4);
  float*  u     = (float*) alloc((size_t)M_TOT * D_ * 4);
  __bf16* ub    = (__bf16*)alloc((size_t)M_TOT * D_ * 2);
  float*  dbc   = (float*) alloc((size_t)M_TOT * (R_ + 2 * N_) * 4);
  float*  delta = (float*) alloc((size_t)M_TOT * D_ * 4);
  __bf16* zb    = (__bf16*)alloc((size_t)M_TOT * D_ * 2);

  auto cvt = [&](const float* src, __bf16* dst, int n) {
    int blocks = (n + 1023) / 1024;
    if (blocks > 4096) blocks = 4096;
    cvt_bf16_kernel<<<blocks, 256, 0, stream>>>(src, dst, n);
  };
  cvt(x, x_bf, M_TOT * D_);
  cvt(W1, w1b, D_ * D_);
  cvt(W2, w2b, D_ * D_);
  cvt(W_dbc, wdbcb, (R_ + 2 * N_) * D_);
  cvt(W_dt, wdtb, D_ * R_);
  cvt(Wout, woutb, D_ * D_);
  pack_convw_kernel<<<(L_ * L_) / 256, 256, 0, stream>>>(conv_w, cwb);

  // x1_pre = x@W1^T + b1  -> transposed-per-batch bf16 (conv B operand)
  gemm_kernel<128, 128, 4, 2, 0, false><<<dim3(D_ / 128, M_TOT / 128), 256, 0, stream>>>(
      x_bf, D_, w1b, D_, b1, nullptr, x1t, D_, D_);
  // g2 = silu(x@W2^T + b2)
  gemm_kernel<128, 128, 4, 2, 1, false><<<dim3(D_ / 128, M_TOT / 128), 256, 0, stream>>>(
      x_bf, D_, w2b, D_, b2, g2, nullptr, D_, D_);
  // conv + bias + silu -> u (fp32 + bf16)
  conv_gemm_kernel<<<dim3(D_ / 128, L_ / 128, B_), 256, 0, stream>>>(
      cwb, x1t, conv_b, u, ub);
  // dbc = u@W_dbc^T   (N = 64)
  gemm_kernel<128, 64, 8, 1, 2, false><<<dim3(1, M_TOT / 128), 256, 0, stream>>>(
      ub, D_, wdbcb, D_, nullptr, dbc, nullptr, D_, R_ + 2 * N_);
  // delta = softplus(dbc[:, :32]@W_dt^T + b_dt)   (K = 32, fp32 A converted on the fly)
  gemm_kernel<128, 128, 4, 2, 4, true><<<dim3(D_ / 128, M_TOT / 128), 256, 0, stream>>>(
      dbc, R_ + 2 * N_, wdtb, R_, b_dt, delta, nullptr, R_, D_);
  // selective scan, fused with gating + skip -> z (bf16)
  scan_kernel<<<dim3(D_ / 128, B_), 128, 0, stream>>>(
      delta, u, dbc, x, g2, A_log, Dp, zb);
  // out = z@Wout^T + bout
  gemm_kernel<128, 128, 4, 2, 3, false><<<dim3(D_ / 128, M_TOT / 128), 256, 0, stream>>>(
      zb, D_, woutb, D_, bout, (float*)d_out, nullptr, D_, D_);
}
